// REDiffPoolLayer_50672024158932
// MI455X (gfx1250) — matrix operand order repaired and checked by reading, not compiled, not run on hardware
//
#include <hip/hip_runtime.h>
#include <math.h>

// Problem constants (match reference)
#define B_   32
#define N_   1024
#define D_   256
#define H_   256
#define E_   256
#define C_   256
#define KTOP 128

typedef __attribute__((ext_vector_type(16))) __bf16 bf16x16;
typedef __attribute__((ext_vector_type(8)))  float  f32x8;
typedef unsigned int  u32x4 __attribute__((ext_vector_type(4)));
typedef int           i32x4 __attribute__((ext_vector_type(4)));
typedef int           i32x8 __attribute__((ext_vector_type(8)));
typedef unsigned short u16;

#if defined(__has_builtin)
#  if __has_builtin(__builtin_amdgcn_tensor_load_to_lds) && \
      __has_builtin(__builtin_amdgcn_s_wait_tensorcnt)
#    define HAVE_TDM 1
#  endif
#endif
#ifndef HAVE_TDM
#  define HAVE_TDM 0
#endif

__device__ __forceinline__ u16 f2bf(float f) {
  unsigned u = __float_as_uint(f);
  u += 0x7FFFu + ((u >> 16) & 1u);           // round-to-nearest-even
  return (u16)(u >> 16);
}

union FragU { bf16x16 v; u16 u[16]; };

// ---------------- WMMA GEMM core (bf16 in, f32 accumulate) ----------------
// Tile: 64x64 output per block (128 threads = 4 waves), K-step 32.
// LDA/LDB are compile-time leading dims (256 or 1024 everywhere in this net)
// so staging loads use immediate offsets: no per-element address ALU/xcnt.
#define TM 64
#define TN 64
#define TK 32
#define LPITCH 40   // u16 pitch; 80 bytes/row keeps 16B alignment of fragments

#if HAVE_TDM
// Issue one TDM descriptor: load a TM x TK bf16 tile (row-major, leading dim
// lda elements) into LDS at lds_byte_off with a 4-DWORD pad after each 64B row
// -> reproduces the LPITCH=40 u16 layout the WMMA fragment gather expects.
__device__ __forceinline__ void tdm_load_a_tile(const u16* gsrc, int lda_elems,
                                                unsigned lds_byte_off)
{
  unsigned long long ga = (unsigned long long)(uintptr_t)gsrc;
  u32x4 g0;
  g0[0] = 1u;                                  // count=1, user-mode
  g0[1] = lds_byte_off;                        // lds_addr (bytes)
  g0[2] = (unsigned)ga;                        // global_addr[31:0]
  g0[3] = (unsigned)((ga >> 32) & 0x01FFFFFFu) | (2u << 30);   // type=2 (image)
  i32x8 g1;
  g1[0] = (int)((1u << 16) | (1u << 20) | (3u << 22) | (3u << 25));
      // data_size=2B | pad_enable | pad_interval=16 DW (one row) | pad_amount=4 DW
  const unsigned td0 = (unsigned)lda_elems;    // tensor_dim0 = row length
  const unsigned td1 = 1u << 24;               // tensor_dim1: tiles always in range
  g1[1] = (int)((td0 & 0xFFFFu) << 16);                         // td0[15:0]
  g1[2] = (int)(((td0 >> 16) & 0xFFFFu) | ((td1 & 0xFFFFu) << 16));
  g1[3] = (int)(((td1 >> 16) & 0xFFFFu) | ((unsigned)TK << 16)); // tile_dim0=32
  g1[4] = (int)TM;                             // tile_dim1=64 rows, tile_dim2=0
  g1[5] = (int)lda_elems;                      // tensor_dim0_stride[31:0]
  g1[6] = 0;                                   // stride hi / dim1 stride
  g1[7] = 0;
  i32x4 gz4; gz4[0] = 0; gz4[1] = 0; gz4[2] = 0; gz4[3] = 0;   // groups 2/3 unused (2D)
  i32x8 gz8;
#pragma unroll
  for (int i = 0; i < 8; ++i) gz8[i] = 0;
  // 6-arg form on this toolchain: (g0, g1, g2, g3, extra, cpol)
  __builtin_amdgcn_tensor_load_to_lds(g0, g1, gz4, gz4, gz8, 0);
}
#endif

template <int LDA, int LDB>
__device__ __forceinline__ void gemm_tile(
    const u16* __restrict__ A,
    const u16* __restrict__ Bm,
    int Kdim, int m0g, int n0g, int tid,
    f32x8 acc[4], u16* Alds, u16* Blds)
{
  const int wave = tid >> 5, lane = tid & 31;
  const int am  = lane & 15;
  const int akb = (lane >> 4) << 4;   // A: upper half-wave holds K 16..31
  const int bn  = lane & 15;
  const int bkb = (lane >> 4) << 4;   // B: upper half-wave holds K 16..31
#if HAVE_TDM
  const unsigned a_lds_off = (unsigned)(uintptr_t)Alds;  // aperture low bits = LDS byte offset
#endif

  for (int k0 = 0; k0 < Kdim; k0 += TK) {
    __syncthreads();   // previous tile fully consumed

#if HAVE_TDM
    if ((tid >> 5) == 0)   // one wave issues the A-tile DMA (EXEC ignored by TDM)
      tdm_load_a_tile(A + (long long)m0g * LDA + k0, LDA, a_lds_off);
#else
    {   // manual A staging: batch loads into regs, then store (one loadcnt wait)
      const int k  = tid & 31;
      const int mm = tid >> 5;                // 0..3
      const u16* pA = A + (long long)(m0g + mm) * LDA + (k0 + k);
      u16* pS = Alds + mm * LPITCH + k;
      u16 tmpa[16];
#pragma unroll
      for (int i = 0; i < 16; ++i) tmpa[i] = pA[4 * LDA * i];   // imm offsets
#pragma unroll
      for (int i = 0; i < 16; ++i) pS[4 * i * LPITCH] = tmpa[i];
    }
#endif

    {   // manual B staging with transpose: per-thread fixed column n.
        // Compile-time LDB -> all 16 loads share one address VGPR with
        // immediate offsets (no address adds, no xcnt stalls), single wait.
      const int n  = tid & 63;
      const int kk = tid >> 6;                // 0 or 1
      const u16* pB = Bm + (long long)(k0 + kk) * LDB + (n0g + n);
      u16* pS = Blds + n * LPITCH + kk;
      u16 tmpb[16];
#pragma unroll
      for (int i = 0; i < 16; ++i) tmpb[i] = pB[2 * LDB * i];   // imm offsets
#pragma unroll
      for (int i = 0; i < 16; ++i) pS[2 * i] = tmpb[i];
    }

    if (k0 + TK < Kdim) {   // gfx1250 global_prefetch_b8 of next K-tiles
      __builtin_prefetch(Bm + (long long)(k0 + TK + (tid >> 6)) * LDB + n0g + (tid & 63), 0, 1);
#if !HAVE_TDM
      __builtin_prefetch(A + (long long)(m0g + (tid & 63)) * LDA + (k0 + TK), 0, 1);
#endif
    }

#if HAVE_TDM
    if ((tid >> 5) == 0) __builtin_amdgcn_s_wait_tensorcnt(0);
#endif
    __syncthreads();   // publish LDS tile (TDM data + manual stores)

    // A fragment: 16x32 bf16 per wave, contiguous K (half-wave h holds k 16h..16h+15)
    FragU fa;
    const u16* pa = Alds + (wave * 16 + am) * LPITCH + akb;
#pragma unroll
    for (int j = 0; j < 8; ++j) { fa.u[j] = pa[j]; fa.u[8 + j] = pa[8 + j]; }

#pragma unroll
    for (int t = 0; t < 4; ++t) {
      FragU fb;   // B fragment: 32x16 bf16, lane = column, contiguous K
      const u16* pb = Blds + (t * 16 + bn) * LPITCH + bkb;
#pragma unroll
      for (int j = 0; j < 16; ++j) fb.u[j] = pb[j];
      acc[t] = __builtin_amdgcn_wmma_f32_16x16x32_bf16(
          false, fa.v, false, fb.v, (short)0, acc[t], false, false);
    }
  }
}

// Generic batched GEMM: D = A*B (+ A2*B2) (+bias) (*rowscale) -> f32 and/or bf16
template <int LDA, int LDB>
__global__ void __launch_bounds__(128) k_gemm(
    const u16* A,  long long sA,
    const u16* Bm, long long sB,
    const u16* A2, long long sA2,
    const u16* B2, long long sB2,
    int K1, int K2,
    float* Df, long long sDf,
    u16*   Db, long long sDb, int ldd,
    const float* bias,
    const float* rowscale, long long sRS)
{
  __shared__ u16 Alds[TM * LPITCH];
  __shared__ u16 Blds[TN * LPITCH];
  const int tid = threadIdx.x;
  const int b   = blockIdx.z;
  const int m0g = blockIdx.x * TM, n0g = blockIdx.y * TN;

  f32x8 acc[4];
#pragma unroll
  for (int t = 0; t < 4; ++t)
#pragma unroll
    for (int r = 0; r < 8; ++r) acc[t][r] = 0.0f;

  gemm_tile<LDA, LDB>(A + (long long)b * sA, Bm + (long long)b * sB,
                      K1, m0g, n0g, tid, acc, Alds, Blds);
  if (A2)
    gemm_tile<LDA, LDB>(A2 + (long long)b * sA2, B2 + (long long)b * sB2,
                        K2, m0g, n0g, tid, acc, Alds, Blds);

  const int lane = tid & 31, wave = tid >> 5;
  const int rbase = m0g + wave * 16 + ((lane >> 4) << 3);
#pragma unroll
  for (int t = 0; t < 4; ++t) {
    const int col = n0g + t * 16 + (lane & 15);
    const float bv = bias ? bias[col] : 0.0f;
#pragma unroll
    for (int r = 0; r < 8; ++r) {
      const int row = rbase + r;
      float v = acc[t][r] + bv;
      if (rowscale) v *= rowscale[(long long)b * sRS + row];
      if (Df) Df[(long long)b * sDf + (long long)row * ldd + col] = v;
      if (Db) Db[(long long)b * sDb + (long long)row * ldd + col] = f2bf(v);
    }
  }
}

// Fused ||R - A*B||^2 tile accumulation (link / recon losses; never materialize NxN)
template <int LDA, int LDB>
__global__ void __launch_bounds__(128) k_gemm_sqdiff(
    const u16* A,  long long sA,
    const u16* Bm, long long sB,
    int K1,
    const float* R, long long sR, int ldr,
    float* accum, int slot)
{
  __shared__ u16 Alds[TM * LPITCH];
  __shared__ u16 Blds[TN * LPITCH];
  __shared__ float red[128];
  const int tid = threadIdx.x;
  const int b   = blockIdx.z;
  const int m0g = blockIdx.x * TM, n0g = blockIdx.y * TN;

  f32x8 acc[4];
#pragma unroll
  for (int t = 0; t < 4; ++t)
#pragma unroll
    for (int r = 0; r < 8; ++r) acc[t][r] = 0.0f;

  gemm_tile<LDA, LDB>(A + (long long)b * sA, Bm + (long long)b * sB,
                      K1, m0g, n0g, tid, acc, Alds, Blds);

  const int lane = tid & 31, wave = tid >> 5;
  const int rbase = m0g + wave * 16 + ((lane >> 4) << 3);
  float part = 0.0f;
#pragma unroll
  for (int t = 0; t < 4; ++t) {
    const int col = n0g + t * 16 + (lane & 15);
#pragma unroll
    for (int r = 0; r < 8; ++r) {
      const int row = rbase + r;
      float d = R[(long long)b * sR + (long long)row * ldr + col] - acc[t][r];
      part += d * d;
    }
  }
  red[tid] = part; __syncthreads();
  for (int s = 64; s > 0; s >>= 1) {
    if (tid < s) red[tid] += red[tid + s];
    __syncthreads();
  }
  if (tid == 0) atomicAdd(&accum[slot], red[0]);
}

// ---------------- elementwise / row kernels ----------------
__global__ void k_f2bf(const float* src, u16* dst, long long n) {
  long long i = (long long)blockIdx.x * blockDim.x + threadIdx.x;
  const long long stride = (long long)gridDim.x * blockDim.x;
  for (; i < n; i += stride) dst[i] = f2bf(src[i]);
}

__global__ void k_rdeg(const float* adj, float* rdeg) {
  __shared__ float red[128];
  const long long row = blockIdx.x;
  float sum = 0.0f;
  for (int i = threadIdx.x; i < N_; i += 128) sum += adj[row * N_ + i];
  red[threadIdx.x] = sum; __syncthreads();
  for (int s = 64; s > 0; s >>= 1) {
    if (threadIdx.x < s) red[threadIdx.x] += red[threadIdx.x + s];
    __syncthreads();
  }
  if (threadIdx.x == 0) rdeg[row] = 1.0f / fmaxf(red[0], 1.0f);
}

__global__ void k_l2relu(const float* pre, u16* h_bf) {
  __shared__ float red[H_];
  const long long row = blockIdx.x;
  const int c = threadIdx.x;
  const float v = pre[row * H_ + c];
  red[c] = v * v; __syncthreads();
  for (int s = 128; s > 0; s >>= 1) {
    if (c < s) red[c] += red[c + s];
    __syncthreads();
  }
  const float nrm = fmaxf(sqrtf(red[0]), 1e-12f);
  h_bf[row * H_ + c] = f2bf(fmaxf(v / nrm, 0.0f));
}

__global__ void k_softmax(const float* logits, float* s_raw) {
  __shared__ float red[C_];
  const long long row = blockIdx.x;
  const int c = threadIdx.x;
  const float v = logits[row * C_ + c];
  red[c] = v; __syncthreads();
  for (int s = 128; s > 0; s >>= 1) {
    if (c < s) red[c] = fmaxf(red[c], red[c + s]);
    __syncthreads();
  }
  const float m = red[0]; __syncthreads();
  const float e = expf(v - m);
  red[c] = e; __syncthreads();
  for (int s = 128; s > 0; s >>= 1) {
    if (c < s) red[c] += red[c + s];
    __syncthreads();
  }
  s_raw[row * C_ + c] = e / red[0];
}

__global__ void k_scores(const float* s_raw, float* scores) {
  const int b = blockIdx.x, c = threadIdx.x;
  const float* p = s_raw + (long long)b * N_ * C_ + c;
  float sum = 0.0f;
  for (int n = 0; n < N_; ++n) sum += p[(long long)n * C_];
  scores[b * C_ + c] = sum;
}

// top-K threshold by rank counting (C=256, O(C^2) in LDS) + sigmoid gate
__global__ void k_gate(const float* scores, float* gate) {
  __shared__ float sc[C_];
  __shared__ float th;
  const int b = blockIdx.x, c = threadIdx.x;
  sc[c] = scores[b * C_ + c];
  __syncthreads();
  const float v = sc[c];
  int g = 0, ge = 0;
  for (int j = 0; j < C_; ++j) { g += (sc[j] > v); ge += (sc[j] >= v); }
  if (g <= KTOP - 1 && ge >= KTOP) th = v;   // K-th largest (ties benign)
  __syncthreads();
  gate[b * C_ + c] = 1.0f / (1.0f + expf(-(v - th) * 10.0f));  // 1/TEMP = 10
}

__global__ void k_apply_gate(const float* s_raw, const float* gate,
                             float* s_out, u16* s_bf, u16* sT_bf) {
  long long i = (long long)blockIdx.x * blockDim.x + threadIdx.x;
  const long long total = (long long)B_ * N_ * C_;
  const long long stride = (long long)gridDim.x * blockDim.x;
  for (; i < total; i += stride) {
    const int c = (int)(i & (C_ - 1));
    const long long bn = i >> 8;
    const int n = (int)(bn & (N_ - 1));
    const int b = (int)(bn >> 10);
    const float v = s_raw[i] * gate[b * C_ + c];
    s_out[i] = v;
    const u16 h = f2bf(v);
    s_bf[i] = h;
    sT_bf[((long long)b * C_ + c) * N_ + n] = h;
  }
}

__global__ void k_ent(const float* s, float* accum) {
  __shared__ float red[256];
  float part = 0.0f;
  const long long total = (long long)B_ * N_ * C_;
  for (long long i = (long long)blockIdx.x * 256 + threadIdx.x; i < total;
       i += (long long)gridDim.x * 256) {
    const float v = s[i];
    part += -v * logf(v + 1e-15f);
  }
  red[threadIdx.x] = part; __syncthreads();
  for (int st = 128; st > 0; st >>= 1) {
    if (threadIdx.x < st) red[threadIdx.x] += red[threadIdx.x + st];
    __syncthreads();
  }
  if (threadIdx.x == 0) atomicAdd(&accum[1], red[0]);
}

__global__ void k_clu(const float* sts, float* accum) {
  __shared__ float red[256];
  const int b = blockIdx.x, t = threadIdx.x;
  const float* p = sts + (long long)b * C_ * C_;
  float ss = 0.0f;
  for (int i = t; i < C_ * C_; i += 256) { const float v = p[i]; ss += v * v; }
  red[t] = ss; __syncthreads();
  for (int st = 128; st > 0; st >>= 1) {
    if (t < st) red[t] += red[t + st];
    __syncthreads();
  }
  const float nrm = fmaxf(sqrtf(red[0]), 1e-12f);
  __syncthreads();
  float acc = 0.0f;
  for (int i = t; i < C_ * C_; i += 256) {
    const int r = i >> 8, c2 = i & 255;
    const float d = p[i] / nrm - ((r == c2) ? 0.0625f : 0.0f);  // I/sqrt(256)
    acc += d * d;
  }
  red[t] = acc; __syncthreads();
  for (int st = 128; st > 0; st >>= 1) {
    if (t < st) red[t] += red[t + st];
    __syncthreads();
  }
  if (t == 0) atomicAdd(&accum[2], sqrtf(red[0]));
}

__global__ void k_finalize(const float* accum, float* out4) {
  if (threadIdx.x == 0) {
    const float inv_numel = 1.0f / 33554432.0f;      // B*N*N
    out4[0] = sqrtf(accum[0]) * inv_numel;           // link
    out4[1] = accum[1] * (1.0f / (float)(B_ * N_));  // ent
    out4[2] = accum[2] * (1.0f / (float)B_);         // clu
    out4[3] = sqrtf(accum[3]) * inv_numel;           // recon
  }
}

// ---------------- host orchestration ----------------
extern "C" void kernel_launch(void* const* d_in, const int* in_sizes, int n_in,
                              void* d_out, int out_size, void* d_ws, size_t ws_size,
                              hipStream_t stream)
{
  (void)in_sizes; (void)n_in; (void)out_size; (void)ws_size;
  const float* x    = (const float*)d_in[0];
  const float* adj  = (const float*)d_in[1];
  const float* Wr_p = (const float*)d_in[2];
  const float* br_p = (const float*)d_in[3];
  const float* Wo_p = (const float*)d_in[4];
  const float* Wl_p = (const float*)d_in[5];
  const float* bl_p = (const float*)d_in[6];
  const float* Wr_e = (const float*)d_in[7];
  const float* br_e = (const float*)d_in[8];
  const float* Wo_e = (const float*)d_in[9];
  const float* Wl_e = (const float*)d_in[10];
  const float* bl_e = (const float*)d_in[11];

  float* out = (float*)d_out;
  const long long XP_OFF    = 0;
  const long long ADJP_OFF  = (long long)B_ * C_ * E_;
  const long long S_OFF     = ADJP_OFF + (long long)B_ * C_ * C_;
  const long long LINK_OFF  = S_OFF + (long long)B_ * N_ * C_;
  const long long NODEX_OFF = LINK_OFF + 4;

  size_t cur = 0;
  auto alloc = [&](size_t bytes) -> void* {
    void* p = (char*)d_ws + cur;
    cur += (bytes + 255) & ~(size_t)255;
    return p;
  };
  u16*   xb      = (u16*)  alloc((size_t)B_ * N_ * D_ * 2);
  u16*   adjb    = (u16*)  alloc((size_t)B_ * N_ * N_ * 2);
  float* rdeg    = (float*)alloc((size_t)B_ * N_ * 4);
  u16*   aggb    = (u16*)  alloc((size_t)B_ * N_ * D_ * 2);
  u16*   wrp_b   = (u16*)  alloc((size_t)D_ * H_ * 2);
  u16*   wop_b   = (u16*)  alloc((size_t)D_ * H_ * 2);
  u16*   wlp_b   = (u16*)  alloc((size_t)H_ * C_ * 2);
  u16*   wre_b   = (u16*)  alloc((size_t)D_ * H_ * 2);
  u16*   woe_b   = (u16*)  alloc((size_t)D_ * H_ * 2);
  u16*   wle_b   = (u16*)  alloc((size_t)H_ * E_ * 2);
  float* pre     = (float*)alloc((size_t)B_ * N_ * H_ * 4);
  u16*   h_bf    = (u16*)  alloc((size_t)B_ * N_ * H_ * 2);
  float* logits  = (float*)alloc((size_t)B_ * N_ * C_ * 4);
  float* s_raw   = (float*)alloc((size_t)B_ * N_ * C_ * 4);
  float* scores  = (float*)alloc((size_t)B_ * C_ * 4);
  float* gate    = (float*)alloc((size_t)B_ * C_ * 4);
  u16*   s_bf    = (u16*)  alloc((size_t)B_ * N_ * C_ * 2);
  u16*   sT_bf   = (u16*)  alloc((size_t)B_ * C_ * N_ * 2);
  u16*   nodex_bf= (u16*)  alloc((size_t)B_ * N_ * E_ * 2);
  u16*   sTa_bf  = (u16*)  alloc((size_t)B_ * C_ * N_ * 2);
  u16*   adjp_bf = (u16*)  alloc((size_t)B_ * C_ * C_ * 2);
  float* sts     = (float*)alloc((size_t)B_ * C_ * C_ * 4);
  u16*   t_bf    = (u16*)  alloc((size_t)B_ * N_ * C_ * 2);
  float* accum   = (float*)alloc(4 * sizeof(float));

  (void)hipMemsetAsync(accum, 0, 4 * sizeof(float), stream);

  // Convert inputs/weights to bf16
  k_f2bf<<<4096, 256, 0, stream>>>(x,   xb,   (long long)B_ * N_ * D_);
  k_f2bf<<<8192, 256, 0, stream>>>(adj, adjb, (long long)B_ * N_ * N_);
  k_f2bf<<<64, 256, 0, stream>>>(Wr_p, wrp_b, (long long)D_ * H_);
  k_f2bf<<<64, 256, 0, stream>>>(Wo_p, wop_b, (long long)D_ * H_);
  k_f2bf<<<64, 256, 0, stream>>>(Wl_p, wlp_b, (long long)H_ * C_);
  k_f2bf<<<64, 256, 0, stream>>>(Wr_e, wre_b, (long long)D_ * H_);
  k_f2bf<<<64, 256, 0, stream>>>(Wo_e, woe_b, (long long)D_ * H_);
  k_f2bf<<<64, 256, 0, stream>>>(Wl_e, wle_b, (long long)H_ * E_);
  k_rdeg<<<B_ * N_, 128, 0, stream>>>(adj, rdeg);

  const u16* nul = nullptr;
  // agg = (adj @ x) / deg  -> bf16      (A: lda=1024, B: ldb=256)
  k_gemm<1024, 256><<<dim3(N_/TM, D_/TN, B_), 128, 0, stream>>>(
      adjb, (long long)N_*N_,  xb, (long long)N_*D_,
      nul, 0, nul, 0, N_, 0,
      nullptr, 0, aggb, (long long)N_*D_, D_, nullptr, rdeg, N_);

  // ---- pool branch ----
  k_gemm<256, 256><<<dim3(N_/TM, H_/TN, B_), 128, 0, stream>>>(
      aggb, (long long)N_*D_,  wrp_b, 0,
      xb,   (long long)N_*D_,  wop_b, 0,  D_, D_,
      pre, (long long)N_*H_, nullptr, 0, H_, br_p, nullptr, 0);
  k_l2relu<<<B_ * N_, H_, 0, stream>>>(pre, h_bf);
  k_gemm<256, 256><<<dim3(N_/TM, C_/TN, B_), 128, 0, stream>>>(
      h_bf, (long long)N_*H_,  wlp_b, 0,
      nul, 0, nul, 0, H_, 0,
      logits, (long long)N_*C_, nullptr, 0, C_, bl_p, nullptr, 0);

  // ---- embed branch (node_x goes straight to output + bf16 copy) ----
  k_gemm<256, 256><<<dim3(N_/TM, H_/TN, B_), 128, 0, stream>>>(
      aggb, (long long)N_*D_,  wre_b, 0,
      xb,   (long long)N_*D_,  woe_b, 0,  D_, D_,
      pre, (long long)N_*H_, nullptr, 0, H_, br_e, nullptr, 0);
  k_l2relu<<<B_ * N_, H_, 0, stream>>>(pre, h_bf);
  k_gemm<256, 256><<<dim3(N_/TM, E_/TN, B_), 128, 0, stream>>>(
      h_bf, (long long)N_*H_,  wle_b, 0,
      nul, 0, nul, 0, H_, 0,
      out + NODEX_OFF, (long long)N_*E_, nodex_bf, (long long)N_*E_, E_,
      bl_e, nullptr, 0);

  // softmax -> scores -> soft top-K gate -> gated S (f32 out + bf16 + bf16^T)
  k_softmax<<<B_ * N_, C_, 0, stream>>>(logits, s_raw);
  k_scores<<<B_, C_, 0, stream>>>(s_raw, scores);
  k_gate<<<B_, C_, 0, stream>>>(scores, gate);
  k_apply_gate<<<4096, 256, 0, stream>>>(s_raw, gate, out + S_OFF, s_bf, sT_bf);
  k_ent<<<2048, 256, 0, stream>>>(out + S_OFF, accum);

  // xp = S^T @ node_x   (A=sT lda=1024, B ldb=256)
  k_gemm<1024, 256><<<dim3(C_/TM, E_/TN, B_), 128, 0, stream>>>(
      sT_bf, (long long)C_*N_,  nodex_bf, (long long)N_*E_,
      nul, 0, nul, 0, N_, 0,
      out + XP_OFF, (long long)C_*E_, nullptr, 0, E_, nullptr, nullptr, 0);

  // sTa = S^T @ adj  (bf16)   (lda=1024, ldb=1024)
  k_gemm<1024, 1024><<<dim3(C_/TM, N_/TN, B_), 128, 0, stream>>>(
      sT_bf, (long long)C_*N_,  adjb, (long long)N_*N_,
      nul, 0, nul, 0, N_, 0,
      nullptr, 0, sTa_bf, (long long)C_*N_, N_, nullptr, nullptr, 0);

  // adj_p = sTa @ S  (f32 output + bf16 copy)   (lda=1024, ldb=256)
  k_gemm<1024, 256><<<dim3(C_/TM, C_/TN, B_), 128, 0, stream>>>(
      sTa_bf, (long long)C_*N_,  s_bf, (long long)N_*C_,
      nul, 0, nul, 0, N_, 0,
      out + ADJP_OFF, (long long)C_*C_, adjp_bf, (long long)C_*C_, C_,
      nullptr, nullptr, 0);

  // sts = S^T @ S -> cluster loss   (lda=1024, ldb=256)
  k_gemm<1024, 256><<<dim3(C_/TM, C_/TN, B_), 128, 0, stream>>>(
      sT_bf, (long long)C_*N_,  s_bf, (long long)N_*C_,
      nul, 0, nul, 0, N_, 0,
      sts, (long long)C_*C_, nullptr, 0, C_, nullptr, nullptr, 0);
  k_clu<<<B_, 256, 0, stream>>>(sts, accum);

  // link = ||adj - S S^T||_F (fused)   (A=s lda=256, B=sT ldb=1024)
  k_gemm_sqdiff<256, 1024><<<dim3(N_/TM, N_/TN, B_), 128, 0, stream>>>(
      s_bf, (long long)N_*C_,  sT_bf, (long long)C_*N_, C_,
      adj, (long long)N_*N_, N_, accum, 0);

  // t = S @ adj_p ; recon = ||adj - t S^T||_F (fused)
  k_gemm<256, 256><<<dim3(N_/TM, C_/TN, B_), 128, 0, stream>>>(
      s_bf, (long long)N_*C_,  adjp_bf, (long long)C_*C_,
      nul, 0, nul, 0, C_, 0,
      nullptr, 0, t_bf, (long long)N_*C_, C_, nullptr, nullptr, 0);
  k_gemm_sqdiff<256, 1024><<<dim3(N_/TM, N_/TN, B_), 128, 0, stream>>>(
      t_bf, (long long)N_*C_,  sT_bf, (long long)C_*N_, C_,
      adj, (long long)N_*N_, N_, accum, 3);

  k_finalize<<<1, 32, 0, stream>>>(accum, out + LINK_OFF);
}
